// MultiHeadAttention_62070867362186
// MI455X (gfx1250) — matrix operand order repaired and checked by reading, not compiled, not run on hardware
//
#include <hip/hip_runtime.h>

typedef _Float16 h16;
typedef __attribute__((ext_vector_type(16))) _Float16 v16h;
typedef __attribute__((ext_vector_type(8)))  _Float16 v8h;
typedef __attribute__((ext_vector_type(4)))  _Float16 v4h;
typedef __attribute__((ext_vector_type(8)))  float    v8f;
typedef __attribute__((ext_vector_type(4)))  float    f32x4;
typedef __attribute__((ext_vector_type(4)))  int      i32x4;
typedef __attribute__((address_space(1))) i32x4 gas_i32x4;
typedef __attribute__((address_space(3))) i32x4 las_i32x4;

// Problem constants
constexpr int Bsz = 4;
constexpr int Seq = 2048;
constexpr int Dm  = 1024;
constexpr int Hn  = 16;
constexpr int Hd  = 64;
constexpr int Mrows = Bsz * Seq;   // 8192

// ---------------------------------------------------------------------------
// CDNA5 async global->LDS copy (16B per lane), guarded for toolchain support.
// Signature (probe-confirmed): (global v4i*, local v4i*, imm offset, imm cpol)
// ---------------------------------------------------------------------------
#if __has_builtin(__builtin_amdgcn_global_load_async_to_lds_b128)
#define HAVE_ASYNC_COPY 1
#else
#define HAVE_ASYNC_COPY 0
#endif

static __device__ inline void async_copy16(h16* lds_dst, const h16* gsrc) {
#if HAVE_ASYNC_COPY
  __builtin_amdgcn_global_load_async_to_lds_b128(
      (gas_i32x4*)(i32x4*)(void*)const_cast<h16*>(gsrc),
      (las_i32x4*)(i32x4*)(void*)lds_dst, 0, 0);
#else
  *(f32x4*)lds_dst = *(const f32x4*)gsrc;
#endif
}

static __device__ inline void wait_async0() {
#if HAVE_ASYNC_COPY
#if __has_builtin(__builtin_amdgcn_s_wait_asynccnt)
  __builtin_amdgcn_s_wait_asynccnt(0);
#else
  asm volatile("s_wait_asynccnt 0" ::: "memory");
#endif
#endif
}

// ---------------------------------------------------------------------------
// WMMA fragment loaders (wave32, 16x16x32 f16, per CDNA5 ISA VGPR layouts).
// All loads are contiguous 16B vector LDS loads -> ds_load_b128.
// ---------------------------------------------------------------------------
static __device__ inline v16h load_a_frag(const h16* base, int ld, int row0, int k0) {
  const int lane = threadIdx.x & 31;
  const int hf   = lane >> 4;
  const int r    = row0 + (lane & 15);
  const h16* p   = base + r * ld + k0 + 8 * hf;
  v8h lo = *(const v8h*)p;
  v8h hi = *(const v8h*)(p + 16);
  return __builtin_shufflevector(lo, hi, 0,1,2,3,4,5,6,7,8,9,10,11,12,13,14,15);
}

// B (32x16 KxN) gathered from the TRANSPOSED store T[n][k] (row-major, ld elems).
static __device__ inline v16h load_b_fragT(const h16* base, int ld, int k0, int col0) {
  const int lane = threadIdx.x & 31;
  const int hf   = lane >> 4;
  const int n    = col0 + (lane & 15);
  const h16* p   = base + n * ld + k0 + 8 * hf;
  v8h lo = *(const v8h*)p;
  v8h hi = *(const v8h*)(p + 16);
  return __builtin_shufflevector(lo, hi, 0,1,2,3,4,5,6,7,8,9,10,11,12,13,14,15);
}

static __device__ inline v8f wmma16(const v16h& a, const v16h& b, const v8f& c) {
  return __builtin_amdgcn_wmma_f32_16x16x32_f16(false, a, false, b, (short)0, c, false, false);
}

// ---------------------------------------------------------------------------
// Kernel A: QKV projection. Y = X @ W + bias, f16 out, head-split [B*H, S, Hd].
// grid = (D/128, M/128, 3), block = 256 (8 waves, 2x4). TK = 64.
// Single-buffered: 38KB LDS -> ~8 blocks/WGP -> 16 waves/SIMD hides latency.
// ---------------------------------------------------------------------------
constexpr int TM = 128, TN = 128, TK = 64;
constexpr int LDA  = TK + 8;    // 72 elems -> 144B rows (16B aligned frags)
constexpr int LDBT = TK + 16;   // 80 elems -> 160B rows (16B aligned frags)

__global__ __launch_bounds__(256) void qkv_proj_f16_kernel(
    const float* __restrict__ qin, const float* __restrict__ kin, const float* __restrict__ vin,
    const float* __restrict__ Wq,  const float* __restrict__ Wk,  const float* __restrict__ Wv,
    const float* __restrict__ bq,  const float* __restrict__ bk,  const float* __restrict__ bv,
    h16* __restrict__ Qh, h16* __restrict__ Kh, h16* __restrict__ Vh)
{
  const int which = blockIdx.z;
  const float* X    = (which == 0) ? qin : (which == 1) ? kin : vin;
  const float* W    = (which == 0) ? Wq  : (which == 1) ? Wk  : Wv;
  const float* bias = (which == 0) ? bq  : (which == 1) ? bk  : bv;
  h16*         Y    = (which == 0) ? Qh  : (which == 1) ? Kh  : Vh;

  __shared__ h16 As[TM][LDA];     // activations, row-major [m][k]
  __shared__ h16 BsT[TN][LDBT];   // weights, TRANSPOSED [n][k]

  const int tid  = threadIdx.x;
  const int wave = tid >> 5;
  const int wm   = wave >> 2;
  const int wn   = wave & 3;
  const int m_base = blockIdx.y * TM;
  const int n_base = blockIdx.x * TN;

  v8f acc[4][2] = {};

  for (int k0 = 0; k0 < Dm; k0 += TK) {
    for (int i = tid; i < TM * TK / 4; i += 256) {
      int r  = i >> 4;
      int c4 = (i & 15) << 2;
      f32x4 x = *(const f32x4*)&X[(m_base + r) * Dm + k0 + c4];
      v4h hx = { (h16)x.x, (h16)x.y, (h16)x.z, (h16)x.w };
      *(v4h*)&As[r][c4] = hx;
    }
    for (int i = tid; i < TK * TN / 4; i += 256) {
      int r  = i >> 5;
      int c4 = (i & 31) << 2;
      f32x4 w = *(const f32x4*)&W[(k0 + r) * Dm + n_base + c4];
      BsT[c4 + 0][r] = (h16)w.x;
      BsT[c4 + 1][r] = (h16)w.y;
      BsT[c4 + 2][r] = (h16)w.z;
      BsT[c4 + 3][r] = (h16)w.w;
    }
    __syncthreads();

#pragma unroll
    for (int ks = 0; ks < TK; ks += 32) {
      v16h af[4], bf[2];
#pragma unroll
      for (int i = 0; i < 4; ++i) af[i] = load_a_frag(&As[0][0], LDA, wm * 64 + i * 16, ks);
#pragma unroll
      for (int j = 0; j < 2; ++j) bf[j] = load_b_fragT(&BsT[0][0], LDBT, ks, wn * 32 + j * 16);
#pragma unroll
      for (int i = 0; i < 4; ++i)
#pragma unroll
        for (int j = 0; j < 2; ++j)
          acc[i][j] = wmma16(af[i], bf[j], acc[i][j]);
    }
    __syncthreads();
  }

  const int lane = tid & 31;
  const int hf   = lane >> 4;
  const int nl   = lane & 15;
#pragma unroll
  for (int i = 0; i < 4; ++i)
#pragma unroll
    for (int j = 0; j < 2; ++j)
#pragma unroll
      for (int vv = 0; vv < 8; ++vv) {
        int gm = m_base + wm * 64 + i * 16 + vv + 8 * hf;
        int gn = n_base + wn * 32 + j * 16 + nl;
        float val = acc[i][j][vv] + bias[gn];
        int b = gm >> 11, s = gm & (Seq - 1);
        int h = gn >> 6,  d = gn & (Hd - 1);
        Y[(((size_t)(b * Hn + h)) * Seq + s) * Hd + d] = (h16)val;
      }
}

// ---------------------------------------------------------------------------
// Kernel B: flash attention per (b,h). block = 128 (4 waves); wave w owns rows
// [w*16, +16) of the 64-row Q tile. grid = (S/64, B*H).
// Double-buffered K/V tiles: next tile's async K copy + V global loads are in
// flight while current tile's WMMA/softmax runs.
// ---------------------------------------------------------------------------
constexpr int QT = 64, KT = 64;
constexpr int LQ  = Hd + 8;    // 72: A-frag source + async dst (144B rows)
constexpr int LK  = Hd + 16;   // 80: B-frag source + async dst (160B rows)
constexpr int LVT = KT + 16;   // 80: transposed V [d][key]
constexpr int LSS = KT + 4;    // 68 floats
constexpr int LPP = KT + 8;    // 72: A-frag source
constexpr int NKT = Seq / KT;  // 32 KV tiles

__global__ __launch_bounds__(128) void flash_attn_f16_kernel(
    const h16* __restrict__ Qh, const h16* __restrict__ Kh,
    const h16* __restrict__ Vh, h16* __restrict__ Ctx)
{
  const int bh = blockIdx.y;
  const int q0 = blockIdx.x * QT;
  const size_t base = (size_t)bh * Seq * Hd;
  const h16* Qp = Qh + base;
  const h16* Kp = Kh + base;
  const h16* Vp = Vh + base;
  h16*       Cp = Ctx + base;

  __shared__ h16   Qs[QT][LQ];
  __shared__ h16   Ks[2][KT][LK];     // [key][d] (transposed B for Q@K^T)
  __shared__ h16   VsT[2][Hd][LVT];   // [d][key]
  __shared__ float Ss[QT][LSS];
  __shared__ h16   Ps[QT][LPP];
  __shared__ float mrow[QT], lrow[QT], arow[QT];

  const int tid  = threadIdx.x;
  const int wave = tid >> 5;
  const int lane = tid & 31;
  const int hf   = lane >> 4;
  const int nl   = lane & 15;

  // Per-thread copy coordinates: 512 16B-chunks per 64x64 tile, 4 per thread.
  int cr[4], cc[4];
#pragma unroll
  for (int n = 0; n < 4; ++n) {
    int i = tid + n * 128;
    cr[n] = i >> 3;
    cc[n] = (i & 7) << 3;
  }

  // Q tile (async) + first K tile (async) + first V tile (regs -> scatter).
#pragma unroll
  for (int n = 0; n < 4; ++n)
    async_copy16(&Qs[cr[n]][cc[n]], Qp + (size_t)(q0 + cr[n]) * Hd + cc[n]);
#pragma unroll
  for (int n = 0; n < 4; ++n)
    async_copy16(&Ks[0][cr[n]][cc[n]], Kp + (size_t)cr[n] * Hd + cc[n]);
  {
    v8h vr[4];
#pragma unroll
    for (int n = 0; n < 4; ++n)
      vr[n] = *(const v8h*)(Vp + (size_t)cr[n] * Hd + cc[n]);
#pragma unroll
    for (int n = 0; n < 4; ++n)
#pragma unroll
      for (int j = 0; j < 8; ++j) VsT[0][cc[n] + j][cr[n]] = vr[n][j];
  }
  if (tid < QT) { mrow[tid] = -1e30f; lrow[tid] = 0.0f; }

  v8f o[4] = {};
  const float scale = 0.125f;   // 1/sqrt(64)
  wait_async0();
  __syncthreads();

  for (int t = 0; t < NKT; ++t) {
    const int cur = t & 1, nxt = cur ^ 1;
    const int k1 = (t + 1) * KT;

    // Prefetch next tile into the inactive buffers while computing this one.
    v8h vr[4];
    if (t + 1 < NKT) {
#pragma unroll
      for (int n = 0; n < 4; ++n)
        async_copy16(&Ks[nxt][cr[n]][cc[n]], Kp + (size_t)(k1 + cr[n]) * Hd + cc[n]);
#pragma unroll
      for (int n = 0; n < 4; ++n)
        vr[n] = *(const v8h*)(Vp + (size_t)(k1 + cr[n]) * Hd + cc[n]);
    }

    // S = Q @ K^T (wave rows, all 64 key cols), contraction over Hd = 64.
    v8f sacc[4] = {};
#pragma unroll
    for (int ks = 0; ks < Hd; ks += 32) {
      v16h aq = load_a_frag(&Qs[0][0], LQ, wave * 16, ks);
#pragma unroll
      for (int j = 0; j < 4; ++j) {
        v16h bk = load_b_fragT(&Ks[cur][0][0], LK, ks, j * 16);
        sacc[j] = wmma16(aq, bk, sacc[j]);
      }
    }
#pragma unroll
    for (int j = 0; j < 4; ++j)
#pragma unroll
      for (int vv = 0; vv < 8; ++vv) {
        int r = wave * 16 + vv + 8 * hf;
        int c = j * 16 + nl;
        Ss[r][c] = sacc[j][vv] * scale;
      }
    __syncthreads();

    // Online softmax: 2 threads per row, combine via lane-pair shuffle.
    {
      const int r = tid >> 1;
      const int cb = (tid & 1) << 5;   // 0 or 32
      float m_old = mrow[r];
      float pm = m_old;
      for (int c = 0; c < 32; ++c) pm = fmaxf(pm, Ss[r][cb + c]);
      float m = fmaxf(pm, __shfl_xor(pm, 1));
      float lsum = 0.0f;
      for (int c = 0; c < 32; ++c) {
        float p = __expf(Ss[r][cb + c] - m);
        lsum += p;
        Ps[r][cb + c] = (h16)p;
      }
      lsum += __shfl_xor(lsum, 1);
      if ((tid & 1) == 0) {
        float alpha = __expf(m_old - m);
        mrow[r] = m;
        lrow[r] = lrow[r] * alpha + lsum;
        arow[r] = alpha;
      }
    }
    __syncthreads();

    // Rescale running output, then O += P @ V.
#pragma unroll
    for (int j = 0; j < 4; ++j)
#pragma unroll
      for (int vv = 0; vv < 8; ++vv) {
        int r = wave * 16 + vv + 8 * hf;
        o[j][vv] = o[j][vv] * arow[r];
      }
#pragma unroll
    for (int ks = 0; ks < KT; ks += 32) {
      v16h ap = load_a_frag(&Ps[0][0], LPP, wave * 16, ks);
#pragma unroll
      for (int j = 0; j < 4; ++j) {
        v16h bv = load_b_fragT(&VsT[cur][0][0], LVT, ks, j * 16);
        o[j] = wmma16(ap, bv, o[j]);
      }
    }

    // Scatter prefetched V into the inactive buffer (current reads are done
    // in this thread; cross-wave reads only touch [cur]).
    if (t + 1 < NKT) {
#pragma unroll
      for (int n = 0; n < 4; ++n)
#pragma unroll
        for (int j = 0; j < 8; ++j) VsT[nxt][cc[n] + j][cr[n]] = vr[n][j];
    }
    wait_async0();
    __syncthreads();
  }

  // Finalize: O /= l, store context f16 [B*H, S, Hd].
#pragma unroll
  for (int j = 0; j < 4; ++j)
#pragma unroll
    for (int vv = 0; vv < 8; ++vv) {
      int r = wave * 16 + vv + 8 * hf;
      int c = j * 16 + nl;
      float val = o[j][vv] / lrow[r];
      Cp[(size_t)(q0 + r) * Hd + c] = (h16)val;
    }
}

// ---------------------------------------------------------------------------
// Kernel C: output projection. Y = Ctx(head-merge) @ Wo + bo, f32 out.
// grid = (D/128, M/128), block = 256.
// ---------------------------------------------------------------------------
__global__ __launch_bounds__(256) void out_proj_f16_kernel(
    const h16* __restrict__ Ctx, const float* __restrict__ Wo,
    const float* __restrict__ bo, float* __restrict__ Y)
{
  __shared__ h16 As[TM][LDA];
  __shared__ h16 BsT[TN][LDBT];

  const int tid  = threadIdx.x;
  const int wave = tid >> 5;
  const int wm   = wave >> 2;
  const int wn   = wave & 3;
  const int m_base = blockIdx.y * TM;
  const int n_base = blockIdx.x * TN;

  v8f acc[4][2] = {};

  for (int k0 = 0; k0 < Dm; k0 += TK) {
    for (int i = tid; i < TM * TK / 8; i += 256) {
      int r  = i >> 3;
      int c8 = (i & 7) << 3;
      int gm = m_base + r;
      int kk = k0 + c8;
      int b = gm >> 11, s = gm & (Seq - 1);
      int h = kk >> 6,  d = kk & (Hd - 1);
      v8h cx = *(const v8h*)&Ctx[(((size_t)(b * Hn + h)) * Seq + s) * Hd + d];
      *(v8h*)&As[r][c8] = cx;
    }
    for (int i = tid; i < TK * TN / 4; i += 256) {
      int r  = i >> 5;
      int c4 = (i & 31) << 2;
      f32x4 w = *(const f32x4*)&Wo[(k0 + r) * Dm + n_base + c4];
      BsT[c4 + 0][r] = (h16)w.x;
      BsT[c4 + 1][r] = (h16)w.y;
      BsT[c4 + 2][r] = (h16)w.z;
      BsT[c4 + 3][r] = (h16)w.w;
    }
    __syncthreads();

#pragma unroll
    for (int ks = 0; ks < TK; ks += 32) {
      v16h af[4], bf[2];
#pragma unroll
      for (int i = 0; i < 4; ++i) af[i] = load_a_frag(&As[0][0], LDA, wm * 64 + i * 16, ks);
#pragma unroll
      for (int j = 0; j < 2; ++j) bf[j] = load_b_fragT(&BsT[0][0], LDBT, ks, wn * 32 + j * 16);
#pragma unroll
      for (int i = 0; i < 4; ++i)
#pragma unroll
        for (int j = 0; j < 2; ++j)
          acc[i][j] = wmma16(af[i], bf[j], acc[i][j]);
    }
    __syncthreads();
  }

  const int lane = tid & 31;
  const int hf   = lane >> 4;
  const int nl   = lane & 15;
#pragma unroll
  for (int i = 0; i < 4; ++i)
#pragma unroll
    for (int j = 0; j < 2; ++j)
#pragma unroll
      for (int vv = 0; vv < 8; ++vv) {
        int gm = m_base + wm * 64 + i * 16 + vv + 8 * hf;
        int gn = n_base + wn * 32 + j * 16 + nl;
        Y[(size_t)gm * Dm + gn] = acc[i][j][vv] + bo[gn];
      }
}

// ---------------------------------------------------------------------------
// Host launcher
// ---------------------------------------------------------------------------
extern "C" void kernel_launch(void* const* d_in, const int* in_sizes, int n_in,
                              void* d_out, int out_size, void* d_ws, size_t ws_size,
                              hipStream_t stream) {
  const float* q  = (const float*)d_in[0];
  const float* k  = (const float*)d_in[1];
  const float* v  = (const float*)d_in[2];
  const float* Wq = (const float*)d_in[3];
  const float* bq = (const float*)d_in[4];
  const float* Wk = (const float*)d_in[5];
  const float* bk = (const float*)d_in[6];
  const float* Wv = (const float*)d_in[7];
  const float* bv = (const float*)d_in[8];
  const float* Wo = (const float*)d_in[9];
  const float* bo = (const float*)d_in[10];
  float* out = (float*)d_out;

  h16* ws = (h16*)d_ws;
  const size_t chunk = (size_t)Mrows * Dm;   // 8192*1024 f16 elements
  h16* Qh  = ws;
  h16* Kh  = ws + chunk;
  h16* Vh  = ws + 2 * chunk;
  h16* Ctx = ws + 3 * chunk;

  dim3 gA(Dm / TN, Mrows / TM, 3);
  qkv_proj_f16_kernel<<<gA, 256, 0, stream>>>(q, k, v, Wq, Wk, Wv, bq, bk, bv, Qh, Kh, Vh);

  dim3 gB(Seq / QT, Bsz * Hn);
  flash_attn_f16_kernel<<<gB, 128, 0, stream>>>(Qh, Kh, Vh, Ctx);

  dim3 gC(Dm / TN, Mrows / TM);
  out_proj_f16_kernel<<<gC, 256, 0, stream>>>(Ctx, Wo, bo, out);
}
